// EncdecMultiheadAttn_61632780698101
// MI455X (gfx1250) — compile-verified, matrix operand order checked by reading, not run on hardware
//
#include <hip/hip_runtime.h>
#include <hip/hip_bf16.h>

// ---------------------------------------------------------------------------
// EncdecMultiheadAttn for MI455X (gfx1250, wave32, WMMA + TDM).
//
// Shapes: Lq=Lk=2048, B=4, H=16, hd=64, E=1024.
// Outputs: out (2048,4,1024) f32  then coverage (64,2048,2048) f32, flat.
//
// Precision: bf16 WMMA (v_wmma_f32_16x16x32_bf16), f32 accumulate.
// Coverage (1.07GB) is the bandwidth floor (~46us @ 23.3TB/s); scores are
// kept LDS-resident (256KB of the 320KB WGP LDS) so it is written only once.
// Q/K tiles are staged into LDS by the Tensor Data Mover with double-buffered
// ping-pong on TENSORcnt so the DMA for tile kb+1 overlaps WMMA on tile kb.
// ---------------------------------------------------------------------------

#define LQ    2048
#define LKK   2048
#define BSZ   4
#define NH    16
#define EMB   1024
#define HD    64
#define MROWS (LQ * BSZ)   // 8192 rows for all projections
#define KV_N  (2 * EMB)    // 2048 output cols of the kv projection

typedef __attribute__((ext_vector_type(16))) __bf16 v16bf;
typedef __attribute__((ext_vector_type(8)))  float  v8f;
typedef __attribute__((ext_vector_type(4)))  unsigned int v4u;
typedef __attribute__((ext_vector_type(8)))  int v8i;
typedef __attribute__((ext_vector_type(4)))  int v4i;

union FragU { v16bf v; unsigned long long d[4]; };

__device__ __forceinline__ unsigned short f32_bf16(float f) {
  unsigned u = __float_as_uint(f);
  u += 0x7FFFu + ((u >> 16) & 1u);          // round-to-nearest-even
  return (unsigned short)(u >> 16);
}
__device__ __forceinline__ float bf16_f32(unsigned short h) {
  return __uint_as_float(((unsigned)h) << 16);
}

// Load a 16x32 bf16 A/B fragment from an LDS tile stored K-major.
// ISA layout (CDNA5 ISA): lanes 0-15 hold row=lane, elems {K 0-7, 16-23};
// lanes 16-31 hold row=lane-16, elems {K 8-15, 24-31}. Each half is four
// contiguous 4-element (64-bit) groups -> lowers to ds_load_b128 pairs.
__device__ __forceinline__ v16bf ld_frag(const unsigned short* tile, int row,
                                         int ld, int koff) {
  const unsigned long long* q =
      (const unsigned long long*)(tile + row * ld + koff);
  const int h2 = ((threadIdx.x & 31) >> 4) << 1;   // upper half: +8 elems = +2 groups
  FragU u;
  u.d[0] = q[h2 + 0];   // K  0..3  (+8 if upper half)
  u.d[1] = q[h2 + 1];   // K  4..7
  u.d[2] = q[h2 + 4];   // K 16..19
  u.d[3] = q[h2 + 5];   // K 20..23
  return u.v;
}

__device__ __forceinline__ v8f wmma_bf16(v16bf a, v16bf b, v8f c) {
  // (neg_a, A, neg_b, B, c_mod, C, reuse_a, reuse_b)
  return __builtin_amdgcn_wmma_f32_16x16x32_bf16(false, a, false, b,
                                                 (short)0, c, false, false);
}

// ---------------------------------------------------------------------------
// Tensor Data Mover: 2D bf16 tile (tile_h rows x tile_w elems, row pitch
// row_stride_elems) global -> LDS (packed row-major, contiguous).
// D# layout per CDNA5 ISA.
// Issue from ONE wave per workgroup (TDM ignores EXEC); completion tracked
// on TENSORcnt (s_wait_tensorcnt).
// ---------------------------------------------------------------------------
__device__ __forceinline__ void tdm_load_2d(unsigned lds_off_bytes,
                                            const void* gsrc,
                                            unsigned tile_w_elems,
                                            unsigned tile_h,
                                            unsigned long long row_stride_elems) {
  const unsigned long long ga = (unsigned long long)gsrc;

  v4u g0 = {};
  g0[0] = 1u;                                         // count=1, user descriptor
  g0[1] = lds_off_bytes;                              // lds_addr
  g0[2] = (unsigned)(ga & 0xFFFFFFFFull);             // global_addr[31:0]
  g0[3] = (unsigned)((ga >> 32) & 0x1FFFFFFull)       // global_addr[56:32]
          | (2u << 30);                               // type = 2 ("image")

  v8i g1 = {};
  g1[0] = (int)(1u << 16);                            // data_size = 1 (2 bytes)
  g1[1] = (int)((tile_w_elems & 0xFFFFu) << 16);      // tensor_dim0[15:0]
  g1[2] = (int)((tile_h & 0xFFFFu) << 16);            // dim0 hi=0 | tensor_dim1 lo
  g1[3] = (int)((tile_w_elems & 0xFFFFu) << 16);      // dim1 hi=0 | tile_dim0
  g1[4] = (int)(tile_h & 0xFFFFu);                    // tile_dim1 | tile_dim2=0
  g1[5] = (int)(unsigned)(row_stride_elems & 0xFFFFFFFFull);      // stride0 lo
  g1[6] = (int)(unsigned)((row_stride_elems >> 32) & 0xFFFFull);  // stride0 hi
  g1[7] = 0;                                          // stride1 (unused, 2D)

  v4i g2 = {};
  v4i g3 = {};
#if __clang_major__ >= 23
  v8i g4 = {};
  __builtin_amdgcn_tensor_load_to_lds(g0, g1, g2, g3, g4, 0);
#else
  __builtin_amdgcn_tensor_load_to_lds(g0, g1, g2, g3, 0);
#endif
}

// ---------------------------------------------------------------------------
// Kernel 1: f32 -> bf16 pack
// ---------------------------------------------------------------------------
__global__ void pack_bf16(const float* __restrict__ src,
                          unsigned short* __restrict__ dst, int n) {
  int i = blockIdx.x * blockDim.x + threadIdx.x;
  int stride = gridDim.x * blockDim.x;
  for (; i < n; i += stride) dst[i] = f32_bf16(src[i]);
}

// ---------------------------------------------------------------------------
// Kernel 2: C[M,N] = A[M,K] * B[N,K]^T   (both operands K-major, bf16)
// Tile 64x64x32, 256 threads = 8 waves arranged 4(M) x 2(N-half); each wave
// owns two 16x16 accumulators. C written as bf16 or f32.
// ---------------------------------------------------------------------------
__global__ void gemm_bf16_nt(const unsigned short* __restrict__ A,
                             const unsigned short* __restrict__ B,
                             void* __restrict__ C,
                             int M, int N, int K, int c_is_f32) {
  __shared__ unsigned short As[64 * 32];
  __shared__ unsigned short Bs[64 * 32];

  const int tid  = threadIdx.x;
  const int lane = tid & 31;
  const int wave = tid >> 5;
  const int wm   = wave & 3;     // 16-row strip
  const int wn   = wave >> 2;    // 32-col half
  const int m0   = blockIdx.y * 64;
  const int n0   = blockIdx.x * 64;

  const int lr = tid >> 2;       // staging row   0..63
  const int lc = (tid & 3) * 8;  // staging col   0,8,16,24 (8 bf16 = b128)

  v8f c0 = {}; v8f c1 = {};

  for (int kt = 0; kt < K; kt += 32) {
    *(uint4*)&As[lr * 32 + lc] =
        *(const uint4*)&A[(size_t)(m0 + lr) * K + kt + lc];
    *(uint4*)&Bs[lr * 32 + lc] =
        *(const uint4*)&B[(size_t)(n0 + lr) * K + kt + lc];
    __syncthreads();
    if (kt + 32 < K) {  // pull next tiles toward L2 (global_prefetch_b8)
      __builtin_prefetch(&A[(size_t)(m0 + lr) * K + kt + 32 + lc], 0, 1);
      __builtin_prefetch(&B[(size_t)(n0 + lr) * K + kt + 32 + lc], 0, 1);
    }
    v16bf a  = ld_frag(As, wm * 16 + (lane & 15), 32, 0);
    v16bf b0 = ld_frag(Bs, wn * 32 +      (lane & 15), 32, 0);
    v16bf b1 = ld_frag(Bs, wn * 32 + 16 + (lane & 15), 32, 0);
    c0 = wmma_bf16(a, b0, c0);
    c1 = wmma_bf16(a, b1, c1);
    __syncthreads();
  }

  const int half = lane >> 4;
  const int cn   = lane & 15;
#pragma unroll
  for (int r = 0; r < 8; ++r) {         // C layout: M = r + 8*half, N = lane&15
    size_t row  = (size_t)(m0 + wm * 16 + r + 8 * half);
    int    col0 = n0 + wn * 32 + cn;
    if (c_is_f32) {
      ((float*)C)[row * N + col0]      = c0[r];
      ((float*)C)[row * N + col0 + 16] = c1[r];
    } else {
      ((unsigned short*)C)[row * N + col0]      = f32_bf16(c0[r]);
      ((unsigned short*)C)[row * N + col0 + 16] = f32_bf16(c1[r]);
    }
  }
}

// ---------------------------------------------------------------------------
// Kernel 3: fused attention for one (batch,head) and a 64-row Q stripe.
//   grid = (Lq/64, B*H), block = 256 (8 waves), dynamic LDS = 288KB:
//     qs    64x64 bf16      8KB   Q tile        (TDM-staged)
//     kvs0/1 64x64 bf16    16KB   K tile ping-pong (TDM) / V tile (manual)
//     sc    64x2048 bf16  256KB   score/probability stripe (LDS-resident)
//     partial max/sum             softmax reductions
// ---------------------------------------------------------------------------
__global__ void attn_fused(const unsigned short* __restrict__ qp,   // (8192,1024) bf16
                           const unsigned short* __restrict__ kvp,  // (8192,2048) bf16
                           const unsigned char* __restrict__ mask,  // (B,Lk) bool
                           float* __restrict__ cov,                 // (B*H,Lq,Lk) f32
                           unsigned short* __restrict__ ctx) {      // (8192,1024) bf16
  extern __shared__ char smem[];
  unsigned short* qs   = (unsigned short*)smem;         // 4096 elems
  unsigned short* kvs0 = qs + 64 * 64;                  // 4096 elems
  unsigned short* kvs1 = kvs0 + 64 * 64;                // 4096 elems
  unsigned short* sc   = kvs1 + 64 * 64;                // 131072 elems
  float* pmax = (float*)(sc + 64 * LKK);                // 64*4
  float* psum = pmax + 256;                             // 64*4

  const int tid  = threadIdx.x;
  const int lane = tid & 31;
  const int wave = tid >> 5;
  const int wm   = wave & 3;
  const int wn   = wave >> 2;
  const int bh   = blockIdx.y;
  const int b    = bh >> 4;
  const int h    = bh & 15;
  const int q0   = blockIdx.x * 64;
  const float scale = 0.125f;  // hd^-0.5

  const unsigned qs_off   = (unsigned)(unsigned long long)(void*)qs;
  const unsigned kvs_off0 = (unsigned)(unsigned long long)(void*)kvs0;
  const unsigned kvs_off1 = (unsigned)(unsigned long long)(void*)kvs1;

  // ---- TDM prologue: Q tile + K tile 0 in flight ----
  if (wave == 0) {
    const unsigned short* qsrc = qp + ((size_t)(q0 * BSZ + b)) * EMB + h * HD;
    tdm_load_2d(qs_off, qsrc, 64, 64, (unsigned long long)BSZ * EMB);
    const unsigned short* ksrc = kvp + ((size_t)b) * KV_N + h * 128;
    tdm_load_2d(kvs_off0, ksrc, 64, 64, (unsigned long long)BSZ * KV_N);
  }

  // ---- phase 1: scores = (Q K^T)*scale, masked, -> LDS stripe ----
  for (int kb = 0; kb < LKK / 64; ++kb) {
    if (wave == 0) __builtin_amdgcn_s_wait_tensorcnt(0);  // tile kb landed
    __syncthreads();  // tile kb visible to all; reads of the other buffer done
    if (wave == 0 && kb + 1 < LKK / 64) {  // DMA kb+1 overlaps compute of kb
      const unsigned short* ksrc =
          kvp + ((size_t)((kb + 1) * 64) * BSZ + b) * KV_N + h * 128;
      tdm_load_2d((kb & 1) ? kvs_off0 : kvs_off1, ksrc, 64, 64,
                  (unsigned long long)BSZ * KV_N);
    }
    const unsigned short* kcur = (kb & 1) ? kvs1 : kvs0;

    v16bf a0 = ld_frag(qs, wm * 16 + (lane & 15), 64, 0);
    v16bf a1 = ld_frag(qs, wm * 16 + (lane & 15), 64, 32);
#pragma unroll
    for (int j = 0; j < 2; ++j) {
      const int nl = wn * 32 + j * 16;
      v16bf b0 = ld_frag(kcur, nl + (lane & 15), 64, 0);
      v16bf b1 = ld_frag(kcur, nl + (lane & 15), 64, 32);
      v8f cacc = {};
      cacc = wmma_bf16(a0, b0, cacc);
      cacc = wmma_bf16(a1, b1, cacc);
      const int col = nl + (lane & 15);
      const int gk  = kb * 64 + col;
      const bool bad = mask[b * LKK + gk] != 0;
#pragma unroll
      for (int r8 = 0; r8 < 8; ++r8) {
        float s = bad ? -1e30f : cacc[r8] * scale;
        const int row = wm * 16 + r8 + 8 * (lane >> 4);
        sc[row * LKK + gk] = f32_bf16(s);
      }
    }
  }
  __syncthreads();

  // ---- phase 2: softmax over the LDS stripe + write coverage once ----
  {
    const int row = tid >> 2;       // 4 threads per row
    const int qtr = tid & 3;        // each scans 512 columns
    unsigned short* sr = sc + row * LKK + qtr * 512;

    float m = -1e30f;
    for (int i = 0; i < 512; ++i) m = fmaxf(m, bf16_f32(sr[i]));
    pmax[row * 4 + qtr] = m;
    __syncthreads();
    const float rm = fmaxf(fmaxf(pmax[row * 4 + 0], pmax[row * 4 + 1]),
                           fmaxf(pmax[row * 4 + 2], pmax[row * 4 + 3]));
    float sum = 0.0f;
    for (int i = 0; i < 512; ++i) sum += __expf(bf16_f32(sr[i]) - rm);
    psum[row * 4 + qtr] = sum;
    __syncthreads();
    const float rs = psum[row * 4 + 0] + psum[row * 4 + 1] +
                     psum[row * 4 + 2] + psum[row * 4 + 3];
    const float rinv = __builtin_amdgcn_rcpf(rs);

    float* cr = cov + ((size_t)bh * LQ + q0 + row) * LKK + qtr * 512;
    for (int i = 0; i < 512; ++i) {
      float p = __expf(bf16_f32(sr[i]) - rm) * rinv;
      cr[i]  = p;                 // coverage: single HBM write
      sr[i]  = f32_bf16(p);       // P stays in LDS for the PV GEMM
    }
  }

  // ---- phase 3: ctx = P @ V (V staged transposed so B-frags are K-major) ----
  v8f c0 = {}; v8f c1 = {};
  for (int kb = 0; kb < LKK / 64; ++kb) {
    __syncthreads();
    {
      const int r = tid >> 2;        // k index within tile
      const int c = (tid & 3) * 16;  // d base
      const unsigned short* src =
          kvp + ((size_t)((kb * 64 + r) * BSZ + b)) * KV_N + h * 128 + 64 + c;
      uint4 v0 = *(const uint4*)&src[0];
      uint4 v1 = *(const uint4*)&src[8];
      const unsigned short* e0 = (const unsigned short*)&v0;
      const unsigned short* e1 = (const unsigned short*)&v1;
#pragma unroll
      for (int i = 0; i < 8; ++i) {  // vs[d][k] transpose
        kvs0[(c + i) * 64 + r]     = e0[i];
        kvs0[(c + 8 + i) * 64 + r] = e1[i];
      }
    }
    __syncthreads();

    v16bf p0  = ld_frag(sc,   wm * 16 + (lane & 15), LKK, kb * 64);
    v16bf p1  = ld_frag(sc,   wm * 16 + (lane & 15), LKK, kb * 64 + 32);
    v16bf v00 = ld_frag(kvs0, wn * 32 +      (lane & 15), 64, 0);
    v16bf v01 = ld_frag(kvs0, wn * 32 +      (lane & 15), 64, 32);
    v16bf v10 = ld_frag(kvs0, wn * 32 + 16 + (lane & 15), 64, 0);
    v16bf v11 = ld_frag(kvs0, wn * 32 + 16 + (lane & 15), 64, 32);
    c0 = wmma_bf16(p0, v00, c0);
    c0 = wmma_bf16(p1, v01, c0);
    c1 = wmma_bf16(p0, v10, c1);
    c1 = wmma_bf16(p1, v11, c1);
  }

  {
    const int half = lane >> 4;
    const int cn   = lane & 15;
#pragma unroll
    for (int r = 0; r < 8; ++r) {
      const int l  = q0 + wm * 16 + r + 8 * half;
      const int d0 = wn * 32 + cn;
      const size_t base = ((size_t)(l * BSZ + b)) * EMB + h * HD;
      ctx[base + d0]      = f32_bf16(c0[r]);
      ctx[base + d0 + 16] = f32_bf16(c1[r]);
    }
  }
}

// ---------------------------------------------------------------------------
// Host: 5 packs, 2 projection GEMMs, fused attention, output GEMM.
// ---------------------------------------------------------------------------
extern "C" void kernel_launch(void* const* d_in, const int* in_sizes, int n_in,
                              void* d_out, int out_size, void* d_ws,
                              size_t ws_size, hipStream_t stream) {
  (void)in_sizes; (void)n_in; (void)out_size; (void)ws_size;

  const float*         q    = (const float*)d_in[0];
  const float*         key  = (const float*)d_in[1];
  const unsigned char* mask = (const unsigned char*)d_in[2];  // bool (B,Lk)
  const float*         wq   = (const float*)d_in[3];
  const float*         wkv  = (const float*)d_in[4];
  const float*         wo   = (const float*)d_in[5];

  unsigned short* p = (unsigned short*)d_ws;
  unsigned short* qbf  = p; p += (size_t)MROWS * EMB;   // 16MB
  unsigned short* kbf  = p; p += (size_t)MROWS * EMB;   // 16MB
  unsigned short* wqb  = p; p += (size_t)EMB * EMB;     //  2MB
  unsigned short* wkvb = p; p += (size_t)KV_N * EMB;    //  4MB
  unsigned short* wob  = p; p += (size_t)EMB * EMB;     //  2MB
  unsigned short* qpr  = p; p += (size_t)MROWS * EMB;   // 16MB
  unsigned short* kvpr = p; p += (size_t)MROWS * KV_N;  // 32MB
  unsigned short* ctxb = p; p += (size_t)MROWS * EMB;   // 16MB  (~104MB total)

  float* outp = (float*)d_out;
  float* cov  = outp + (size_t)MROWS * EMB;

  // 1) pack f32 -> bf16
  pack_bf16<<<dim3(2048), dim3(256), 0, stream>>>(q,   qbf,  MROWS * EMB);
  pack_bf16<<<dim3(2048), dim3(256), 0, stream>>>(key, kbf,  MROWS * EMB);
  pack_bf16<<<dim3(512),  dim3(256), 0, stream>>>(wq,  wqb,  EMB * EMB);
  pack_bf16<<<dim3(1024), dim3(256), 0, stream>>>(wkv, wkvb, KV_N * EMB);
  pack_bf16<<<dim3(512),  dim3(256), 0, stream>>>(wo,  wob,  EMB * EMB);

  // 2) q = query @ wq^T        (8192x1024x1024)
  gemm_bf16_nt<<<dim3(EMB / 64, MROWS / 64), dim3(256), 0, stream>>>(
      qbf, wqb, qpr, MROWS, EMB, EMB, 0);
  // 3) kv = key @ wkv^T        (8192x2048x1024)
  gemm_bf16_nt<<<dim3(KV_N / 64, MROWS / 64), dim3(256), 0, stream>>>(
      kbf, wkvb, kvpr, MROWS, KV_N, EMB, 0);

  // 4) fused attention (288KB dynamic LDS per workgroup -> 1 WG / WGP)
  const int att_smem = (64 * 64 * 3 + 64 * LKK) * 2 + 2 * 256 * 4;
  hipFuncSetAttribute((const void*)attn_fused,
                      hipFuncAttributeMaxDynamicSharedMemorySize, att_smem);
  attn_fused<<<dim3(LQ / 64, BSZ * NH), dim3(256), att_smem, stream>>>(
      qpr, kvpr, mask, cov, ctxb);

  // 5) out = ctx @ wo^T        (8192x1024x1024, f32 output)
  gemm_bf16_nt<<<dim3(EMB / 64, MROWS / 64), dim3(256), 0, stream>>>(
      ctxb, wob, outp, MROWS, EMB, EMB, 1);
}
